// RNNQNetwork_19121194402449
// MI455X (gfx1250) — matrix operand order repaired and checked by reading, not compiled, not run on hardware
//
#include <hip/hip_runtime.h>
#include <hip/hip_bf16.h>

// RNN (GRU) Q-network, fully fused persistent-workgroup scan for gfx1250.
// T=512, B=512, OBS=H=128, A=18. 32 workgroups x 256 threads (8 wave32s).
// - Weights resident in LDS as bf16 (transposed NxK for the WMMA B operand).
// - All GEMMs via v_wmma_f32_16x16x32_bf16; biases folded into the C operand;
//   relu/bf16 conversion applied to accumulator registers (no LDS round-trip).
// - A-fragments hoisted to registers and reused across N-tiles.
// - obs tiles double-buffered in LDS via GLOBAL_LOAD_ASYNC_TO_LDS_B128
//   (ASYNCcnt) so HBM latency overlaps the GEMM phase of each step.

#define T_DIM   512
#define B_DIM   512
#define OBS_DIM 128
#define H_DIM   128
#define A_DIM   18
#define A_PAD   32
#define ROWS    16   // batch rows per workgroup

typedef __bf16 bf16_t;
typedef bf16_t v16bf __attribute__((ext_vector_type(16)));
typedef float  v8f   __attribute__((ext_vector_type(8)));
typedef unsigned int v4u __attribute__((ext_vector_type(4)));

union FragBF { v16bf v; v4u q[2]; };
struct AFrags { FragBF f[4]; };   // 4 K-slabs of 32 -> full K=128 A operand

__device__ __forceinline__ float sigmoidf(float x) {
  return 1.0f / (1.0f + __expf(-x));
}

// Load all 4 A-fragments (16x128 row-major bf16 in LDS) for this lane.
// 16-bit A 16x32 layout: M = lane&15; lanes 0-15 hold K 0..7,16..23 of the slab,
// lanes 16-31 hold K 8..15,24..31; 2 K per VGPR -> two 16B LDS loads per slab.
__device__ __forceinline__ void load_a_frags(AFrags& A, const bf16_t* __restrict__ sA,
                                             int lane) {
  const int half = lane >> 4;
  const int l16  = lane & 15;
#pragma unroll
  for (int kk = 0; kk < 4; ++kk) {
    const bf16_t* pa = sA + l16 * 128 + kk * 32 + half * 8;
    A.f[kk].q[0] = *(const v4u*)(pa);
    A.f[kk].q[1] = *(const v4u*)(pa + 16);
  }
}

// One 16x16 N-tile: C = A[16x128] * B[128x16] + bias(col), Bt transposed [N][128].
__device__ __forceinline__ v8f gemm_tile(const AFrags& A,
                                         const bf16_t* __restrict__ sBt,
                                         int noff, int lane, float bias_col) {
  const int half = lane >> 4;
  const int l16  = lane & 15;
  v8f acc;
#pragma unroll
  for (int v = 0; v < 8; ++v) acc[v] = bias_col;  // col fixed per lane in C/D layout
#pragma unroll
  for (int kk = 0; kk < 4; ++kk) {
    FragBF b;
    const bf16_t* pb = sBt + (noff + l16) * 128 + kk * 32 + half * 8;
    b.q[0] = *(const v4u*)(pb);
    b.q[1] = *(const v4u*)(pb + 16);
    acc = __builtin_amdgcn_wmma_f32_16x16x32_bf16(
        false, A.f[kk].v, false, b.v, (short)0, acc, false, false);
  }
  return acc;
}

__device__ __forceinline__ void store_acc_f32(const v8f& acc, float* __restrict__ sD,
                                              int ldd, int noff, int lane) {
  const int half = lane >> 4;
  const int l16  = lane & 15;
#pragma unroll
  for (int v = 0; v < 8; ++v)
    sD[(v + half * 8) * ldd + noff + l16] = acc[v];
}

// Async copy of one 16x128 f32 obs tile into LDS (32 bytes per thread, 2x b128).
__device__ __forceinline__ void issue_obs_async(const float* __restrict__ obs,
                                                float* __restrict__ s_dst,
                                                int t, int bbase, int tid) {
  const int r = tid >> 4;              // 0..15
  const int c = (tid & 15) * 8;        // 0..120
  const float* g = obs + ((size_t)t * B_DIM + bbase + r) * OBS_DIM + c;
  unsigned int lds0 = (unsigned int)(size_t)(s_dst + tid * 8);
  unsigned long long g0 = (unsigned long long)(size_t)g;
  asm volatile("global_load_async_to_lds_b128 %0, %1, off"
               :: "v"(lds0), "v"(g0) : "memory");
  unsigned int lds1 = lds0 + 16u;
  unsigned long long g1 = g0 + 16ull;
  asm volatile("global_load_async_to_lds_b128 %0, %1, off"
               :: "v"(lds1), "v"(g1) : "memory");
}

__device__ __forceinline__ void wait_async_all() {
  asm volatile("s_wait_asynccnt 0x0" ::: "memory");
}

__global__ __launch_bounds__(256, 1) void rnn_gru_q_kernel(
    const float* __restrict__ hidden, const float* __restrict__ obs,
    const int* __restrict__ dones,
    const float* __restrict__ Wemb, const float* __restrict__ bemb,
    const float* __restrict__ Wi,   const float* __restrict__ bi,
    const float* __restrict__ Wh,   const float* __restrict__ bhn,
    const float* __restrict__ Wout, const float* __restrict__ bout,
    float* __restrict__ out) {
  // ---- LDS (326,272 B total, <= 327,680 limit) ----
  __shared__ bf16_t s_wemb[128 * 128];       // 32 KB  Wemb^T [n][k]
  __shared__ bf16_t s_wi[384 * 128];         // 96 KB  Wi^T
  __shared__ bf16_t s_wh[384 * 128];         // 96 KB  Wh^T
  __shared__ bf16_t s_wout[A_PAD * 128];     //  8 KB  Wout^T (zero-padded cols)
  __shared__ float  s_obsraw[2][ROWS * 128]; // 16 KB  double-buffered async obs
  __shared__ bf16_t s_xbf[ROWS * 128];       //  4 KB  obs bf16
  __shared__ bf16_t s_embbf[ROWS * 128];     //  4 KB  relu(emb) bf16
  __shared__ bf16_t s_hbf[ROWS * 128];       //  4 KB  h bf16
  __shared__ float  s_gi[ROWS * 384];        // 24 KB  (includes bi)
  __shared__ float  s_gh[ROWS * 384];        // 24 KB
  __shared__ float  s_hf[ROWS * 128];        //  8 KB  h f32
  __shared__ float  s_bemb[128];
  __shared__ float  s_bi[384];
  __shared__ float  s_bhn[128];
  __shared__ float  s_bout[A_PAD];

  const int tid   = threadIdx.x;
  const int lane  = tid & 31;
  const int wave  = tid >> 5;
  const int bbase = blockIdx.x * ROWS;

  // ---- one-time staging: transposed bf16 weights, biases, initial h ----
  for (int i = tid; i < 128 * 128; i += 256) {
    int k = i >> 7, n = i & 127;
    s_wemb[n * 128 + k] = (bf16_t)Wemb[i];
  }
  for (int i = tid; i < 128 * 384; i += 256) {
    int k = i / 384, n = i % 384;
    s_wi[n * 128 + k] = (bf16_t)Wi[i];
  }
  for (int i = tid; i < 128 * 384; i += 256) {
    int k = i / 384, n = i % 384;
    s_wh[n * 128 + k] = (bf16_t)Wh[i];
  }
  for (int i = tid; i < A_PAD * 128; i += 256) {
    int n = i >> 7, k = i & 127;
    s_wout[i] = (n < A_DIM) ? (bf16_t)Wout[k * A_DIM + n] : (bf16_t)0.0f;
  }
  for (int i = tid; i < 128; i += 256) s_bemb[i] = bemb[i];
  for (int i = tid; i < 384; i += 256) s_bi[i]   = bi[i];
  for (int i = tid; i < 128; i += 256) s_bhn[i]  = bhn[i];
  for (int i = tid; i < A_PAD; i += 256) s_bout[i] = (i < A_DIM) ? bout[i] : 0.0f;
#pragma unroll
  for (int it = 0; it < 8; ++it) {
    int i = tid + it * 256;
    int r = i >> 7, n = i & 127;
    s_hf[i] = hidden[(bbase + r) * H_DIM + n];
  }

  // prime the async pipeline with tile t=0
  issue_obs_async(obs, s_obsraw[0], 0, bbase, tid);
  __syncthreads();

  float* const q_out = out + B_DIM * H_DIM;  // output layout: [h_final | q_vals]

  // ======================= sequential scan over T =======================
  for (int t = 0; t < T_DIM; ++t) {
    const float* s_obst = s_obsraw[t & 1];

    wait_async_all();   // own wave's tile-t loads done; barrier covers the rest
    __syncthreads();

    // reset-mask h -> f32/bf16; convert obs tile f32 -> bf16
#pragma unroll
    for (int it = 0; it < 8; ++it) {
      int i = tid + it * 256;
      int r = i >> 7;
      float hv = s_hf[i];
      if (dones[t * B_DIM + bbase + r] != 0) hv = 0.0f;
      s_hf[i]  = hv;
      s_hbf[i] = (bf16_t)hv;
      s_xbf[i] = (bf16_t)s_obst[i];
    }
    __syncthreads();

    // kick off t+1 tile into the other buffer; warm GL2 at t+8
    if (t + 1 < T_DIM)
      issue_obs_async(obs, s_obsraw[(t + 1) & 1], t + 1, bbase, tid);
    if (t + 8 < T_DIM) {
      const int r = tid >> 4, c = (tid & 15) * 8;
      __builtin_prefetch(obs + ((size_t)(t + 8) * B_DIM + bbase + r) * OBS_DIM + c, 0, 1);
    }

    // emb = relu(x @ Wemb + b_emb): 8 N-tiles, one per wave; bias in C operand,
    // relu + bf16-convert on the accumulator, store bf16 directly.
    {
      const int noff = wave * 16;
      const int half = lane >> 4;
      const int col  = noff + (lane & 15);
      AFrags Ax;
      load_a_frags(Ax, s_xbf, lane);
      v8f acc = gemm_tile(Ax, s_wemb, noff, lane, s_bemb[col]);
#pragma unroll
      for (int v = 0; v < 8; ++v) {
        float e = acc[v] > 0.0f ? acc[v] : 0.0f;
        s_embbf[(v + half * 8) * 128 + col] = (bf16_t)e;
      }
    }
    __syncthreads();

    // gi = emb @ Wi + bi,  gh = h @ Wh : 24 N-tiles each, 3+3 per wave.
    // A-fragments loaded once per matrix and reused across the 3 N-tiles.
    {
      const int l16 = lane & 15;
      AFrags Ae, Ah;
      load_a_frags(Ae, s_embbf, lane);
      load_a_frags(Ah, s_hbf, lane);
#pragma unroll
      for (int j = 0; j < 3; ++j) {
        const int noff = (wave * 3 + j) * 16;
        v8f ai = gemm_tile(Ae, s_wi, noff, lane, s_bi[noff + l16]);
        store_acc_f32(ai, s_gi, 384, noff, lane);
        v8f ah = gemm_tile(Ah, s_wh, noff, lane, 0.0f);
        store_acc_f32(ah, s_gh, 384, noff, lane);
      }
    }
    __syncthreads();

    // GRU gates (bi already inside s_gi)
#pragma unroll
    for (int it = 0; it < 8; ++it) {
      int i = tid + it * 256;
      int r = i >> 7, n = i & 127;
      const float* gi_r = s_gi + r * 384;
      const float* gh_r = s_gh + r * 384;
      float rg = sigmoidf(gi_r[n]       + gh_r[n]);
      float zg = sigmoidf(gi_r[128 + n] + gh_r[128 + n]);
      float ng = tanhf(gi_r[256 + n] + rg * (gh_r[256 + n] + s_bhn[n]));
      float hv = (1.0f - zg) * ng + zg * s_hf[i];
      s_hf[i]  = hv;
      s_hbf[i] = (bf16_t)hv;
    }
    __syncthreads();

    // q = h_new @ Wout + b_out : 2 N-tiles (N padded to 32), waves 0-1,
    // bias in C operand, results stored straight to HBM.
    if (wave < 2) {
      const int noff = wave * 16;
      const int half = lane >> 4;
      const int col  = noff + (lane & 15);
      AFrags Ah;
      load_a_frags(Ah, s_hbf, lane);
      v8f acc = gemm_tile(Ah, s_wout, noff, lane, s_bout[col]);
      if (col < A_DIM) {
#pragma unroll
        for (int v = 0; v < 8; ++v) {
          int row = v + half * 8;
          q_out[((size_t)t * B_DIM + bbase + row) * A_DIM + col] = acc[v];
        }
      }
    }
    __syncthreads();
  }

  // h_final
#pragma unroll
  for (int it = 0; it < 8; ++it) {
    int i = tid + it * 256;
    int r = i >> 7, n = i & 127;
    out[(bbase + r) * H_DIM + n] = s_hf[i];
  }
}

extern "C" void kernel_launch(void* const* d_in, const int* in_sizes, int n_in,
                              void* d_out, int out_size, void* d_ws, size_t ws_size,
                              hipStream_t stream) {
  (void)in_sizes; (void)n_in; (void)out_size; (void)d_ws; (void)ws_size;
  const float* hidden = (const float*)d_in[0];
  const float* obs    = (const float*)d_in[1];
  const int*   dones  = (const int*)d_in[2];
  const float* Wemb   = (const float*)d_in[3];
  const float* bemb   = (const float*)d_in[4];
  const float* Wi     = (const float*)d_in[5];
  const float* bi     = (const float*)d_in[6];
  const float* Wh     = (const float*)d_in[7];
  const float* bhn    = (const float*)d_in[8];
  const float* Wout   = (const float*)d_in[9];
  const float* bout   = (const float*)d_in[10];
  float* out = (float*)d_out;

  dim3 grid(B_DIM / ROWS);  // 32 workgroups, one 16-row batch slice each
  dim3 block(256);          // 8 wave32s
  rnn_gru_q_kernel<<<grid, block, 0, stream>>>(hidden, obs, dones, Wemb, bemb,
                                               Wi, bi, Wh, bhn, Wout, bout, out);
}
